// CausalAttention_4037269258778
// MI455X (gfx1250) — matrix operand order repaired and checked by reading, not compiled, not run on hardware
//
#include <hip/hip_runtime.h>
#include <hip/hip_bf16.h>
#include <math.h>

// Problem constants (from reference)
#define T_LEN 1024
#define B_SZ  4
#define E_DIM 512
#define H_HD  8
#define D_HD  64
#define P_PRJ 64
#define K_FT  128           // 2*P
#define M_ROWS (T_LEN*B_SZ) // 4096

typedef __attribute__((ext_vector_type(16))) _Float16 v16h;
typedef __attribute__((ext_vector_type(8)))  _Float16 v8h;
typedef __attribute__((ext_vector_type(8)))  float    v8f;

static __device__ __forceinline__ v16h cat8(v8h lo, v8h hi) {
    v16h r;
#pragma unroll
    for (int i = 0; i < 8; ++i) { r[i] = lo[i]; r[i + 8] = hi[i]; }
    return r;
}

// A-fragment (16x32 f16, row-major source, lda in halfs), contiguous k order (same as load_b_frag):
// lane m = lane&15 ; halves e <-> k = kb+e ; kb = (lane>>4)*16
static __device__ __forceinline__ v16h load_a_frag(const _Float16* Arow0, int lda, int k0, int lane) {
    int m  = lane & 15;
    int kb = (lane >> 4) << 4;
    const _Float16* p = Arow0 + (size_t)m * lda + k0 + kb;
    v8h lo = *(const v8h*)p;
    v8h hi = *(const v8h*)(p + 8);
    return cat8(lo, hi);
}

// B-fragment (32x16 f16) where B[k][n] = W[n][k] (W row-major [N][K], ldw in halfs):
// lane n = lane&15 ; halves e <-> k = (lane>>4)*16 + e  -> 16 contiguous halfs per lane
static __device__ __forceinline__ v16h load_b_frag(const _Float16* Wn0, int ldw, int k0, int lane) {
    int n  = lane & 15;
    int kb = (lane >> 4) << 4;
    return *(const v16h*)(Wn0 + (size_t)n * ldw + k0 + kb);
}

static __device__ __forceinline__ v8f wmma_f16(v16h a, v16h b, v8f c) {
    return __builtin_amdgcn_wmma_f32_16x16x32_f16(false, a, false, b, (short)0, c, false, false);
}

// ---------------------------------------------------------------- convert f32 -> f16
__global__ void cvt_f16_kernel(const float* __restrict__ X, _Float16* __restrict__ Y, int n) {
    int i = blockIdx.x * 256 + threadIdx.x;
    if (i < n) Y[i] = (_Float16)X[i];
}

// ---------------------------------------------------------------- GEMM: O[r][n] = sum_k A[r][k]*W[n][k] + bias[n]
// Register-blocked: wave computes 32x64 (2 M-tiles x 4 N-tiles = 8 WMMAs/k-step
// sharing 2 A-frags + 4 B-frags). block = 128 thr (4 waves) -> 32x256 tile.
// grid = (M/32, N/256)
__global__ __launch_bounds__(128) void gemm_bias_f16(const _Float16* __restrict__ A,
                                                     const _Float16* __restrict__ W,
                                                     const float* __restrict__ bias,
                                                     float* __restrict__ O,
                                                     int M, int N, int Kd) {
    int lane = threadIdx.x & 31, wave = threadIdx.x >> 5;
    int row0 = blockIdx.x * 32;
    int n0   = blockIdx.y * 256 + wave * 64;
    const _Float16* Arow0 = A + (size_t)row0 * Kd;
    const _Float16* Arow1 = A + (size_t)(row0 + 16) * Kd;
    v8f acc[2][4];
    {
        v8f z = {};
#pragma unroll
        for (int mt = 0; mt < 2; ++mt)
#pragma unroll
            for (int nt = 0; nt < 4; ++nt) acc[mt][nt] = z;
    }
    for (int k0 = 0; k0 < Kd; k0 += 32) {
        if (k0 + 32 < Kd) {  // warm next k-block (global_prefetch_b8)
            __builtin_prefetch(Arow0 + (size_t)(lane & 15) * Kd + k0 + 32, 0, 3);
            __builtin_prefetch(W + (size_t)(n0 + (lane & 15)) * Kd + k0 + 32, 0, 3);
        }
        v16h a0 = load_a_frag(Arow0, Kd, k0, lane);
        v16h a1 = load_a_frag(Arow1, Kd, k0, lane);
#pragma unroll
        for (int nt = 0; nt < 4; ++nt) {
            v16h b = load_b_frag(W + (size_t)(n0 + nt * 16) * Kd, Kd, k0, lane);
            acc[0][nt] = wmma_f16(a0, b, acc[0][nt]);
            acc[1][nt] = wmma_f16(a1, b, acc[1][nt]);
        }
    }
#pragma unroll
    for (int nt = 0; nt < 4; ++nt) {
        int n = n0 + nt * 16 + (lane & 15);
        float bv = bias[n];
#pragma unroll
        for (int mt = 0; mt < 2; ++mt) {
            int mbase = row0 + mt * 16 + ((lane >> 4) << 3);
#pragma unroll
            for (int vi = 0; vi < 8; ++vi)
                O[(size_t)(mbase + vi) * N + n] = acc[mt][nt][vi] + bv;
        }
    }
}

// ---------------------------------------------------------------- per-(row,head) L2 normalize, f32 -> f16
// grid = M*H blocks, 64 threads; contiguous 64-element group per block
__global__ __launch_bounds__(64) void rownorm_f16(const float* __restrict__ X, _Float16* __restrict__ Y) {
    __shared__ float red[64];
    int rh = blockIdx.x, d = threadIdx.x;
    size_t base = (size_t)rh * 64;
    float v = X[base + d];
    red[d] = v * v;
    __syncthreads();
    for (int s = 32; s > 0; s >>= 1) {
        if (d < s) red[d] += red[d + s];
        __syncthreads();
    }
    float den = fmaxf(sqrtf(red[0]), 1e-6f);
    Y[base + d] = (_Float16)(v / den);
}

// ---------------------------------------------------------------- phi: proj = Xn @ RM[h]^T ; write sin/cos features
// grid = (M/16, H), block 128 (4 waves = 4 p-tiles of 16)
__global__ __launch_bounds__(128) void phi_kernel(const _Float16* __restrict__ Xn,   // [M][H*D] normalized
                                                  const _Float16* __restrict__ RM,   // [H][P][D]
                                                  _Float16* __restrict__ PHI,        // [M][H][K]
                                                  _Float16* __restrict__ PHIT) {     // [B][H][K][T] or null
    int lane = threadIdx.x & 31, wave = threadIdx.x >> 5;
    int row0 = blockIdx.x * 16;
    int h    = blockIdx.y;
    int p0   = wave * 16;
    const _Float16* Arow = Xn + (size_t)row0 * (H_HD * D_HD) + h * D_HD;
    const _Float16* Brow = RM + ((size_t)h * P_PRJ + p0) * D_HD;
    v8f acc = {};
#pragma unroll
    for (int k0 = 0; k0 < D_HD; k0 += 32) {
        v16h a = load_a_frag(Arow, H_HD * D_HD, k0, lane);
        v16h b = load_b_frag(Brow, D_HD, k0, lane);
        acc = wmma_f16(a, b, acc);
    }
    int p = p0 + (lane & 15);
    int mbase = row0 + ((lane >> 4) << 3);
#pragma unroll
    for (int vi = 0; vi < 8; ++vi) {
        int r = mbase + vi;
        float pr = acc[vi];
        float sv = __sinf(pr) * 0.125f;   // P^-0.5 = 1/8
        float cv = __cosf(pr) * 0.125f;
        size_t ob = ((size_t)r * H_HD + h) * K_FT;
        PHI[ob + p]          = (_Float16)sv;
        PHI[ob + P_PRJ + p]  = (_Float16)cv;
        if (PHIT) {
            int t = r >> 2, b = r & 3;  // r = t*B + b
            size_t tb = (size_t)(b * H_HD + h) * K_FT;
            PHIT[(tb + p)         * T_LEN + t] = (_Float16)sv;
            PHIT[(tb + P_PRJ + p) * T_LEN + t] = (_Float16)cv;
        }
    }
}

// ---------------------------------------------------------------- V transpose: [T,B,E] f32 -> [B,H,D,T] f16
__global__ void vtrans_kernel(const float* __restrict__ V, _Float16* __restrict__ VT) {
    int idx = blockIdx.x * 256 + threadIdx.x;
    if (idx >= M_ROWS * E_DIM) return;
    int r = idx >> 9, c = idx & 511;
    int t = r >> 2, b = r & 3, h = c >> 6, d = c & 63;
    VT[((size_t)((b * H_HD + h) * D_HD + d) << 10) + t] = (_Float16)V[idx];
}

// ---------------------------------------------------------------- chunked linear-attention scan
// one block per (b,h); 128 thr = 4 waves; wave w owns d-tile [16w,16w+16)
// state ST[d][k] (f32 master in WMMA C fragments, f16 mirror in LDS), z[k] in LDS
__global__ __launch_bounds__(128) void scan_kernel(const _Float16* __restrict__ PQ,   // [T,B,H,K]
                                                   const _Float16* __restrict__ PK,   // [T,B,H,K]
                                                   const _Float16* __restrict__ PKT,  // [B,H,K,T]
                                                   const _Float16* __restrict__ VT,   // [B,H,D,T]
                                                   _Float16* __restrict__ ATTNH) {    // [T,B,E] f16
    __shared__ _Float16 ShT[D_HD][K_FT];   // 16 KB: S^T in f16
    __shared__ _Float16 sc[16][32];        // masked scores, zero-padded to kt=32
    __shared__ float z_lds[K_FT];
    __shared__ float qz_lds[16];
    __shared__ float rs_lds[16];

    int tid = threadIdx.x, lane = tid & 31, wave = tid >> 5;
    int bh = blockIdx.x, b = bh >> 3, h = bh & 7;
    int dt = wave * 16;

    for (int i = tid; i < D_HD * K_FT; i += 128) ((_Float16*)ShT)[i] = (_Float16)0.f;
    for (int i = tid; i < 16 * 32; i += 128)     ((_Float16*)sc)[i]  = (_Float16)0.f;
    z_lds[tid] = 0.f;
    v8f st[8];
    {
        v8f z = {};
#pragma unroll
        for (int nt = 0; nt < 8; ++nt) st[nt] = z;
    }
    __syncthreads();

    const size_t bhK_T = (size_t)(b * H_HD + h) * K_FT * T_LEN;
    const size_t bhD_T = (size_t)(b * H_HD + h) * D_HD * T_LEN;

    for (int tc = 0; tc < T_LEN / 16; ++tc) {
        int t0 = tc * 16;

        // Pq A-fragments for this chunk (shared by inter + scores)
        v16h aq[4];
        {
            int m = lane & 15, kb = (lane >> 4) << 4;
            const _Float16* base = PQ + (((size_t)(t0 + m) * B_SZ + b) * H_HD + h) * K_FT;
#pragma unroll
            for (int ks = 0; ks < 4; ++ks) {
                v8h lo = *(const v8h*)(base + ks * 32 + kb);
                v8h hi = *(const v8h*)(base + ks * 32 + kb + 8);
                aq[ks] = cat8(lo, hi);
            }
        }

        // inter = Pq @ S_prev  (B from LDS S^T: lane n=d, contiguous k)
        v8f acc = {};
        {
            int n = lane & 15, kb = (lane >> 4) << 4;
#pragma unroll
            for (int ks = 0; ks < 4; ++ks) {
                v16h bsh = *(const v16h*)(&ShT[dt + n][ks * 32 + kb]);
                acc = wmma_f16(aq[ks], bsh, acc);
            }
        }

        if (wave == 0) {
            // scores = Pq @ Pk^T (B: lane n = key-time, contiguous over k-features)
            v8f scf = {};
            int n = lane & 15, kb = (lane >> 4) << 4;
#pragma unroll
            for (int ks = 0; ks < 4; ++ks) {
                const _Float16* bp = PK + (((size_t)(t0 + n) * B_SZ + b) * H_HD + h) * K_FT + ks * 32 + kb;
                scf = wmma_f16(aq[ks], *(const v16h*)bp, scf);
            }
            // causal mask (j <= i), write f16 scores to LDS, rowsum via xor-shuffle over 16 lanes
            int j = lane & 15;
            float rv[8];
#pragma unroll
            for (int vi = 0; vi < 8; ++vi) {
                int i = vi + ((lane >> 4) << 3);
                float val = (j <= i) ? scf[vi] : 0.f;
                sc[i][j] = (_Float16)val;
                rv[vi] = val;
            }
#pragma unroll
            for (int off = 1; off < 16; off <<= 1)
#pragma unroll
                for (int vi = 0; vi < 8; ++vi) rv[vi] += __shfl_xor(rv[vi], off);
            if ((lane & 15) == 0) {
                int ib = (lane >> 4) << 3;
#pragma unroll
                for (int vi = 0; vi < 8; ++vi) rs_lds[ib + vi] = rv[vi];
            }
        } else if (wave == 1) {
            // qz_inter[i] = pq_i . z_prev   (split K over lane halves, combine via shfl)
            int i = lane & 15, hf = lane >> 4;
            const _Float16* pqr = PQ + (((size_t)(t0 + i) * B_SZ + b) * H_HD + h) * K_FT + hf * 64;
            float dot = 0.f;
            for (int k = 0; k < 64; ++k) dot += (float)pqr[k] * z_lds[hf * 64 + k];
            dot += __shfl_xor(dot, 16);
            if (lane < 16) qz_lds[i] = dot;
        }
        __syncthreads();

        // z += chunk-sum of pk  (waves 2,3; safe: z_prev already consumed)
        if (wave >= 2) {
            int k = tid - 64;
#pragma unroll
            for (int rep = 0; rep < 2; ++rep) {
                int kk = k + rep * 64;
                const _Float16* col = PKT + bhK_T + (size_t)kk * T_LEN + t0;
                float s = 0.f;
#pragma unroll
                for (int e = 0; e < 16; ++e) s += (float)col[e];
                z_lds[kk] += s;
            }
        }

        // intra = tril(scores) @ V_chunk (A from LDS zero-padded to K=32; B kt>=16 lanes zero)
        {
            int m = lane & 15, kb = (lane >> 4) << 4;
            v8h lo = *(const v8h*)(&sc[m][kb]);
            v8h hi = *(const v8h*)(&sc[m][kb + 8]);   // zero region for upper lane half
            v16h asc = cat8(lo, hi);
            v16h bvv = {};
            if (lane < 16)
                bvv = *(const v16h*)(VT + bhD_T + (size_t)(dt + (lane & 15)) * T_LEN + t0);
            acc = wmma_f16(asc, bvv, acc);
        }

        // out = (inter + intra) / max(qz, eps) -> attn f16 [T,B,E]
        {
            int n = lane & 15, ib = (lane >> 4) << 3;
#pragma unroll
            for (int vi = 0; vi < 8; ++vi) {
                int i = ib + vi;
                float qz = qz_lds[i] + rs_lds[i];
                float o = acc[vi] / fmaxf(qz, 1e-6f);
                ATTNH[((size_t)(t0 + i) * B_SZ + b) * E_DIM + h * D_HD + dt + n] = (_Float16)o;
            }
        }

        // state update: S^T += V_chunk^T @ Pk_chunk  (K=16 zero-padded to 32)
        {
            int m = lane & 15;
            v8h lo = *(const v8h*)(VT + bhD_T + (size_t)(dt + m) * T_LEN + t0);
            v8h hi = *(const v8h*)(VT + bhD_T + (size_t)(dt + m) * T_LEN + t0 + 8);
            v8h zz = {};
            v16h aV = (lane < 16) ? cat8(lo, hi) : cat8(zz, zz);
#pragma unroll
            for (int nt = 0; nt < 8; ++nt) {
                v16h bk = {};
                if (lane < 16)
                    bk = *(const v16h*)(PKT + bhK_T + (size_t)(nt * 16 + (lane & 15)) * T_LEN + t0);
                st[nt] = wmma_f16(aV, bk, st[nt]);
            }
            // mirror updated state to LDS f16 for next chunk's inter
            int ib = (lane >> 4) << 3;
#pragma unroll
            for (int nt = 0; nt < 8; ++nt)
#pragma unroll
                for (int vi = 0; vi < 8; ++vi)
                    ShT[dt + ib + vi][nt * 16 + (lane & 15)] = (_Float16)st[nt][vi];
        }
        __syncthreads();
    }
}

// ----------------------------------------------------------------
extern "C" void kernel_launch(void* const* d_in, const int* in_sizes, int n_in,
                              void* d_out, int out_size, void* d_ws, size_t ws_size,
                              hipStream_t stream) {
    const float* x  = (const float*)d_in[0];
    const float* rm = (const float*)d_in[1];
    const float* Wq = (const float*)d_in[2];
    const float* bq = (const float*)d_in[3];
    const float* Wk = (const float*)d_in[4];
    const float* bk = (const float*)d_in[5];
    const float* Wv = (const float*)d_in[6];
    const float* bv = (const float*)d_in[7];
    const float* Wo = (const float*)d_in[8];
    const float* bo = (const float*)d_in[9];
    float* out = (float*)d_out;

    char* ws = (char*)d_ws;
    size_t off = 0;
    auto carve = [&](size_t bytes) -> void* {
        void* p = ws + off;
        off = (off + bytes + 255) & ~(size_t)255;
        return p;
    };

    _Float16* xh   = (_Float16*)carve((size_t)M_ROWS * E_DIM * 2);
    _Float16* Whq  = (_Float16*)carve((size_t)E_DIM * E_DIM * 2);
    _Float16* Whk  = (_Float16*)carve((size_t)E_DIM * E_DIM * 2);
    _Float16* Whv  = (_Float16*)carve((size_t)E_DIM * E_DIM * 2);
    _Float16* Who  = (_Float16*)carve((size_t)E_DIM * E_DIM * 2);
    _Float16* rmh  = (_Float16*)carve((size_t)H_HD * P_PRJ * D_HD * 2);
    float*    qf   = (float*)carve((size_t)M_ROWS * E_DIM * 4);
    float*    kf   = (float*)carve((size_t)M_ROWS * E_DIM * 4);
    float*    vf   = (float*)carve((size_t)M_ROWS * E_DIM * 4);
    _Float16* qnh  = (_Float16*)carve((size_t)M_ROWS * E_DIM * 2);
    _Float16* knh  = (_Float16*)carve((size_t)M_ROWS * E_DIM * 2);
    _Float16* phiq = (_Float16*)carve((size_t)M_ROWS * H_HD * K_FT * 2);
    _Float16* phik = (_Float16*)carve((size_t)M_ROWS * H_HD * K_FT * 2);
    _Float16* phikT= (_Float16*)carve((size_t)B_SZ * H_HD * K_FT * T_LEN * 2);
    _Float16* vT   = (_Float16*)carve((size_t)B_SZ * H_HD * D_HD * T_LEN * 2);
    _Float16* attnh= (_Float16*)carve((size_t)M_ROWS * E_DIM * 2);

    const int NE  = M_ROWS * E_DIM;   // 2,097,152
    const int NW  = E_DIM * E_DIM;    //   262,144
    const int NRM = H_HD * P_PRJ * D_HD;

    cvt_f16_kernel<<<(NE + 255) / 256, 256, 0, stream>>>(x, xh, NE);
    cvt_f16_kernel<<<(NW + 255) / 256, 256, 0, stream>>>(Wq, Whq, NW);
    cvt_f16_kernel<<<(NW + 255) / 256, 256, 0, stream>>>(Wk, Whk, NW);
    cvt_f16_kernel<<<(NW + 255) / 256, 256, 0, stream>>>(Wv, Whv, NW);
    cvt_f16_kernel<<<(NW + 255) / 256, 256, 0, stream>>>(Wo, Who, NW);
    cvt_f16_kernel<<<(NRM + 255) / 256, 256, 0, stream>>>(rm, rmh, NRM);

    dim3 ggrid(M_ROWS / 32, E_DIM / 256);
    gemm_bias_f16<<<ggrid, 128, 0, stream>>>(xh, Whq, bq, qf, M_ROWS, E_DIM, E_DIM);
    gemm_bias_f16<<<ggrid, 128, 0, stream>>>(xh, Whk, bk, kf, M_ROWS, E_DIM, E_DIM);
    gemm_bias_f16<<<ggrid, 128, 0, stream>>>(xh, Whv, bv, vf, M_ROWS, E_DIM, E_DIM);

    rownorm_f16<<<M_ROWS * H_HD, 64, 0, stream>>>(qf, qnh);
    rownorm_f16<<<M_ROWS * H_HD, 64, 0, stream>>>(kf, knh);

    dim3 pgrid(M_ROWS / 16, H_HD);
    phi_kernel<<<pgrid, 128, 0, stream>>>(qnh, rmh, phiq, nullptr);
    phi_kernel<<<pgrid, 128, 0, stream>>>(knh, rmh, phik, phikT);

    vtrans_kernel<<<(NE + 255) / 256, 256, 0, stream>>>(vf, vT);

    scan_kernel<<<B_SZ * H_HD, 128, 0, stream>>>(phiq, phik, phikT, vT, attnh);

    gemm_bias_f16<<<ggrid, 128, 0, stream>>>(attnh, Who, bo, out, M_ROWS, E_DIM, E_DIM);
}
